// LinearAttentionBlock_56100862820442
// MI455X (gfx1250) — matrix operand order repaired and checked by reading, not compiled, not run on hardware
//
#include <hip/hip_runtime.h>
#include <cstdint>
#include <cstddef>

// ---------------------------------------------------------------------------
// LinearAttentionBlock for MI455X (gfx1250).
// All GEMMs via v_wmma_f32_16x16x32_bf16 (fp32 accumulate), fp32->bf16 once.
// Two tile configs: 256x128 (64x64/wave, 1 ds_load per WMMA) for wide GEMMs,
// 128x128 for skinny N=128 GEMMs; split-K + deterministic reduce for attn@V.
// ---------------------------------------------------------------------------

typedef __attribute__((ext_vector_type(16))) __bf16   bf16x16;
typedef __attribute__((ext_vector_type(8)))  float    f32x8;
typedef __attribute__((ext_vector_type(4)))  uint32_t u32x4;

union FragU { u32x4 u[2]; bf16x16 v; };

__device__ __forceinline__ unsigned short f2bf(float f) {
  uint32_t u = __builtin_bit_cast(uint32_t, f);
  u += 0x7FFFu + ((u >> 16) & 1u);            // round-to-nearest-even
  return (unsigned short)(u >> 16);
}

__global__ void f32_to_bf16_kernel(const float* __restrict__ in,
                                   unsigned short* __restrict__ out, int n) {
  int i = blockIdx.x * blockDim.x + threadIdx.x;
  int stride = gridDim.x * blockDim.x;
  for (; i < n; i += stride) out[i] = f2bf(in[i]);
}

// Deterministic split-K reduction: out_bf16[i] = sum_s parts[s*n + i]
__global__ void reduce_splitk_kernel(const float* __restrict__ parts,
                                     unsigned short* __restrict__ out,
                                     int n, int S) {
  int i = blockIdx.x * blockDim.x + threadIdx.x;
  int stride = gridDim.x * blockDim.x;
  for (; i < n; i += stride) {
    float acc = 0.0f;
    for (int s = 0; s < S; ++s) acc += parts[(size_t)s * n + i];
    out[i] = f2bf(acc);
  }
}

#define BK   32
#define LSTR 40   // padded LDS row stride (bf16 elems) to spread banks

// C[M,N] = A[M,Kd] @ B[N,Kd]^T (+bias)(+relu)(+residual)   A,B bf16 row-major.
// Block tile BMv x BNv, 8 waves, each wave MT*16 x NT*16 via 16x16x32 WMMA.
// SPLITK: blockIdx.z selects a K-chunk of length ksplit; writes fp32 partials.
template<int BMv, int BNv, int MT, int NT,
         bool HAS_BIAS, bool RELU, bool RESIDUAL, bool OUT_BF16, bool TRANS_OUT,
         bool SPLITK>
__global__ __launch_bounds__(256)
void gemm_bf16_wmma(const unsigned short* __restrict__ A,
                    const unsigned short* __restrict__ B,
                    const float* __restrict__ bias,
                    const float* __restrict__ residual,
                    void* __restrict__ Cout,
                    int M, int N, int Kd, int ldc, int ksplit) {
  constexpr int WAVES_N = BNv / (NT * 16);
  constexpr int ACH = (BMv * 4) / 256;     // u32x4 chunks per thread for A tile
  constexpr int BCH = (BNv * 4) / 256;     // u32x4 chunks per thread for B tile

  __shared__ alignas(16) unsigned short As[2][BMv * LSTR];
  __shared__ alignas(16) unsigned short Bs[2][BNv * LSTR];

  const int tid  = threadIdx.x;
  const int wave = tid >> 5;
  const int lane = tid & 31;
  const int m0 = blockIdx.y * BMv;
  const int n0 = blockIdx.x * BNv;
  const int wm = (wave / WAVES_N) * (MT * 16);
  const int wn = (wave % WAVES_N) * (NT * 16);

  const int r0 = tid >> 2;             // 0..63
  const int s  = (tid & 3) * 8;        // 0,8,16,24 (bf16 elems)

  int kbase = 0, klen = Kd;
  if (SPLITK) { kbase = blockIdx.z * ksplit; klen = ksplit; }

  const unsigned short* Aab = A + (size_t)m0 * Kd + kbase;
  const unsigned short* Bab = B + (size_t)n0 * Kd + kbase;

  f32x8 acc[MT][NT];
  #pragma unroll
  for (int i = 0; i < MT; ++i)
    #pragma unroll
    for (int j = 0; j < NT; ++j)
      #pragma unroll
      for (int k = 0; k < 8; ++k) acc[i][j][k] = 0.0f;

  u32x4 ra[ACH], rb[BCH];
  auto fetch = [&](int ko) {
    #pragma unroll
    for (int i = 0; i < ACH; ++i)
      ra[i] = *(const u32x4*)(Aab + (size_t)(r0 + i * 64) * Kd + ko + s);
    #pragma unroll
    for (int i = 0; i < BCH; ++i)
      rb[i] = *(const u32x4*)(Bab + (size_t)(r0 + i * 64) * Kd + ko + s);
  };
  auto stage = [&](int buf) {
    #pragma unroll
    for (int i = 0; i < ACH; ++i)
      *(u32x4*)&As[buf][(r0 + i * 64) * LSTR + s] = ra[i];
    #pragma unroll
    for (int i = 0; i < BCH; ++i)
      *(u32x4*)&Bs[buf][(r0 + i * 64) * LSTR + s] = rb[i];
  };

  const int lr = lane & 15;
  const int kb = (lane < 16) ? 0 : 8;    // B frag: lanes 0-15 -> K {0-7,16-23}, 16-31 -> {8-15,24-31}
  const int ka = (lane < 16) ? 0 : 8;    // A frag: lanes 0-15 -> K {0-7,16-23}, 16-31 -> {8-15,24-31}

  const int nk = klen / BK;
  fetch(0);
  stage(0);
  __syncthreads();

  for (int t = 0; t < nk; ++t) {
    const int cur = t & 1;
    if (t + 1 < nk) fetch((t + 1) * BK);
    if (t + 2 < nk) {   // gfx1250 global_prefetch two stages ahead
      __builtin_prefetch(Aab + (size_t)r0 * Kd + (t + 2) * BK + s, 0, 1);
      __builtin_prefetch(Bab + (size_t)r0 * Kd + (t + 2) * BK + s, 0, 1);
    }

    FragU bfr[NT];
    #pragma unroll
    for (int nt = 0; nt < NT; ++nt) {
      const unsigned short* p = &Bs[cur][(wn + nt * 16 + lr) * LSTR + kb];
      bfr[nt].u[0] = *(const u32x4*)p;
      bfr[nt].u[1] = *(const u32x4*)(p + 16);
    }
    #pragma unroll
    for (int mt = 0; mt < MT; ++mt) {
      const unsigned short* p = &As[cur][(wm + mt * 16 + lr) * LSTR + ka];
      FragU af;
      af.u[0] = *(const u32x4*)p;
      af.u[1] = *(const u32x4*)(p + 16);
      #pragma unroll
      for (int nt = 0; nt < NT; ++nt) {
        acc[mt][nt] = __builtin_amdgcn_wmma_f32_16x16x32_bf16(
            false, af.v, false, bfr[nt].v, (short)0, acc[mt][nt], false, false);
      }
    }
    if (t + 1 < nk) stage(cur ^ 1);
    __syncthreads();
  }

  // Epilogue. C layout: lane = column n (lr); VGPR r: row mbase+r (lanes>=16: +8)
  #pragma unroll
  for (int mt = 0; mt < MT; ++mt) {
    #pragma unroll
    for (int nt = 0; nt < NT; ++nt) {
      const int n  = n0 + wn + nt * 16 + lr;
      const int mb = m0 + wm + mt * 16 + ((lane < 16) ? 0 : 8);
      const float bv = HAS_BIAS ? bias[n] : 0.0f;
      #pragma unroll
      for (int r = 0; r < 8; ++r) {
        float v = acc[mt][nt][r] + bv;
        if (RELU) v = fmaxf(v, 0.0f);
        const int m = mb + r;
        if (SPLITK) {
          float* outp = (float*)Cout + (size_t)blockIdx.z * ((size_t)M * ldc);
          outp[(size_t)m * ldc + n] = v;
        } else if (TRANS_OUT) {
          ((unsigned short*)Cout)[(size_t)n * ldc + m] = f2bf(v);
        } else if (OUT_BF16) {
          ((unsigned short*)Cout)[(size_t)m * ldc + n] = f2bf(v);
        } else {
          const size_t idx = (size_t)m * ldc + n;
          const float rr = RESIDUAL ? residual[idx] : 0.0f;
          ((float*)Cout)[idx] = v + rr;
        }
      }
    }
  }
}

// ---------------------------------------------------------------------------

extern "C" void kernel_launch(void* const* d_in, const int* in_sizes, int n_in,
                              void* d_out, int out_size, void* d_ws, size_t ws_size,
                              hipStream_t stream) {
  (void)in_sizes; (void)n_in; (void)out_size; (void)ws_size;
  const int N = 8192, DM = 1024, DK = 128, DFF = 4096;
  const int SK = 8;                         // split-K factor for attn@V
  const int KSPLIT = N / SK;                // 1024

  const float* x  = (const float*)d_in[0];
  const float* Wq = (const float*)d_in[1];
  const float* bq = (const float*)d_in[2];
  const float* Wk = (const float*)d_in[3];
  const float* bk = (const float*)d_in[4];
  const float* Wv = (const float*)d_in[5];
  const float* bv = (const float*)d_in[6];
  const float* Wp = (const float*)d_in[7];
  const float* bp = (const float*)d_in[8];
  const float* W1 = (const float*)d_in[9];
  const float* b1 = (const float*)d_in[10];
  const float* W2 = (const float*)d_in[11];
  const float* b2 = (const float*)d_in[12];

  char* ws = (char*)d_ws;
  size_t off = 0;
  auto alloc = [&](size_t bytes) -> void* {
    void* p = ws + off;
    off += (bytes + 255) & ~(size_t)255;
    return p;
  };
  unsigned short* xb    = (unsigned short*)alloc((size_t)N * DM * 2);
  unsigned short* Wqb   = (unsigned short*)alloc((size_t)DK * DM * 2);
  unsigned short* Wkb   = (unsigned short*)alloc((size_t)DK * DM * 2);
  unsigned short* Wvb   = (unsigned short*)alloc((size_t)DK * DM * 2);
  unsigned short* Wpb   = (unsigned short*)alloc((size_t)DM * DK * 2);
  unsigned short* W1b   = (unsigned short*)alloc((size_t)DFF * DM * 2);
  unsigned short* W2b   = (unsigned short*)alloc((size_t)DM * DFF * 2);
  unsigned short* Qb    = (unsigned short*)alloc((size_t)N * DK * 2);
  unsigned short* Kb    = (unsigned short*)alloc((size_t)N * DK * 2);
  unsigned short* Vtb   = (unsigned short*)alloc((size_t)DK * N * 2);    // V^T [DK, N]
  unsigned short* attnb = (unsigned short*)alloc((size_t)N * N * 2);     // 128 MB
  float*          ctxp  = (float*)alloc((size_t)SK * N * DK * 4);        // split-K partials
  unsigned short* ctxb  = (unsigned short*)alloc((size_t)N * DK * 2);
  float*          x1    = (float*)alloc((size_t)N * DM * 4);             // residual-1 result
  unsigned short* x1b   = (unsigned short*)alloc((size_t)N * DM * 2);
  unsigned short* hb    = (unsigned short*)alloc((size_t)N * DFF * 2);

  auto conv = [&](const float* src, unsigned short* dst, int n) {
    int blocks = (n + 255) / 256;
    if (blocks > 4096) blocks = 4096;
    f32_to_bf16_kernel<<<dim3(blocks), dim3(256), 0, stream>>>(src, dst, n);
  };
  conv(x,  xb,  N * DM);
  conv(Wq, Wqb, DK * DM);
  conv(Wk, Wkb, DK * DM);
  conv(Wv, Wvb, DK * DM);
  conv(Wp, Wpb, DM * DK);
  conv(W1, W1b, DFF * DM);
  conv(W2, W2b, DM * DFF);

  const dim3 blk(256);

  // --- skinny (N=128) GEMMs: 128x128 tile, wave 64x32 ---
  // Q = x @ Wq^T + bq
  gemm_bf16_wmma<128, 128, 4, 2, true, false, false, true, false, false>
      <<<dim3(1, N / 128), blk, 0, stream>>>(xb, Wqb, bq, nullptr, Qb, N, DK, DM, DK, 0);
  // K = x @ Wk^T + bk
  gemm_bf16_wmma<128, 128, 4, 2, true, false, false, true, false, false>
      <<<dim3(1, N / 128), blk, 0, stream>>>(xb, Wkb, bk, nullptr, Kb, N, DK, DM, DK, 0);
  // V^T = (x @ Wv^T + bv)^T  (transposed store, ldc = N)
  gemm_bf16_wmma<128, 128, 4, 2, true, false, false, true, true, false>
      <<<dim3(1, N / 128), blk, 0, stream>>>(xb, Wvb, bv, nullptr, Vtb, N, DK, DM, N, 0);

  // --- attn = Q @ K^T : 256x128 tile, wave 64x64, 2048 workgroups ---
  gemm_bf16_wmma<256, 128, 4, 4, false, false, false, true, false, false>
      <<<dim3(N / 128, N / 256), blk, 0, stream>>>(Qb, Kb, nullptr, nullptr, attnb, N, N, DK, N, 0);

  // --- ctx = attn @ V : split-K (8 x 64 WGs), fp32 partials + reduce ---
  gemm_bf16_wmma<128, 128, 4, 2, false, false, false, false, false, true>
      <<<dim3(1, N / 128, SK), blk, 0, stream>>>(attnb, Vtb, nullptr, nullptr, ctxp,
                                                 N, DK, N, DK, KSPLIT);
  reduce_splitk_kernel<<<dim3(4096), blk, 0, stream>>>(ctxp, ctxb, N * DK, SK);

  // --- x1 = x + ctx @ Wp^T + bp : fp32 out ---
  gemm_bf16_wmma<256, 128, 4, 4, true, false, true, false, false, false>
      <<<dim3(DM / 128, N / 256), blk, 0, stream>>>(ctxb, Wpb, bp, x, x1, N, DM, DK, DM, 0);
  conv(x1, x1b, N * DM);

  // --- h = relu(x1 @ W1^T + b1) ---
  gemm_bf16_wmma<256, 128, 4, 4, true, true, false, true, false, false>
      <<<dim3(DFF / 128, N / 256), blk, 0, stream>>>(x1b, W1b, b1, nullptr, hb, N, DFF, DM, DFF, 0);

  // --- out = x1 + h @ W2^T + b2 : final fp32 output ---
  gemm_bf16_wmma<256, 128, 4, 4, true, false, true, false, false, false>
      <<<dim3(DM / 128, N / 256), blk, 0, stream>>>(hb, W2b, b2, x1, (float*)d_out, N, DM, DFF, DM, 0);
}
